// GAT_12532714569907
// MI455X (gfx1250) — matrix operand order repaired and checked by reading, not compiled, not run on hardware
//
#include <hip/hip_runtime.h>

#define NNODES 100000
#define NEDGES 1600000
#define HEADS  8
#define DHEAD  16
#define FDIM   128   // HEADS*DHEAD = IN_DIM = OUT_DIM

typedef __attribute__((ext_vector_type(16))) _Float16 v16h;
typedef __attribute__((ext_vector_type(8)))  _Float16 v8h;
typedef __attribute__((ext_vector_type(8)))  float    v8f;

// ---- order-preserving float<->uint map for atomic max ----
__device__ __forceinline__ unsigned ford(float f) {
    unsigned u = __float_as_uint(f);
    return (u & 0x80000000u) ? ~u : (u | 0x80000000u);
}
__device__ __forceinline__ float funord(unsigned u) {
    return (u & 0x80000000u) ? __uint_as_float(u & 0x7fffffffu)
                             : __uint_as_float(~u);
}

// ---- pack W [K x 128] f32 row-major into B-fragment order, f16 ----
// Wp[((kb/32)*2 + half)*2048 + n*16 + i] = W[(kb + half*16 + i)*128 + n]
// so each lane's 16 B-fragment halves are one contiguous 32B read.
__global__ void pack_w(const float* __restrict__ W, _Float16* __restrict__ Wp, int K) {
    int t = blockIdx.x * blockDim.x + threadIdx.x;
    if (t >= K * FDIM) return;
    int i  = t & 15;
    int n  = (t >> 4) & 127;
    int sh = t >> 11;            // step*2 + half
    int half = sh & 1, step = sh >> 1;
    int k = step * 32 + half * 16 + i;
    Wp[t] = (_Float16)W[(size_t)k * FDIM + n];
}

__global__ void cvt_f32_to_h(const float* __restrict__ x, _Float16* __restrict__ xh, int n) {
    int t = blockIdx.x * blockDim.x + threadIdx.x;
    if (t < n) xh[t] = (_Float16)x[t];
}

// relu + convert: xh = (f16)max(acc, 0)
__global__ void relu_cvt(const float* __restrict__ acc, _Float16* __restrict__ xh, int n) {
    int t = blockIdx.x * blockDim.x + threadIdx.x;
    if (t < n) { float v = acc[t]; xh[t] = (_Float16)(v > 0.f ? v : 0.f); }
}

// ---- WMMA GEMM: Out[r, 0:128] = [X0h | X1h](r, :) @ W (+bias), K = 128 or 256
// f16 inputs; X rows are length-128 f16; Wp pre-packed per-fragment.
// one wave per 16x16 tile; block = 8 waves = one 16-row stripe, all 128 cols.
template <int K>
__global__ void gemm_wmma_h(const _Float16* __restrict__ X0h,
                            const _Float16* __restrict__ X1h,
                            const _Float16* __restrict__ Wp,
                            const float* __restrict__ bias,
                            float* __restrict__ Out)
{
    const int lane = threadIdx.x & 31;
    const int wv   = threadIdx.x >> 5;     // column tile 0..7 (== head for layer GEMMs)
    const int r0   = blockIdx.x * 16;      // exact: 100000/16 blocks
    const int c0   = wv * 16;
    const int half = lane >> 4;            // 0 or 1
    const int l16  = lane & 15;

    const int arow = r0 + l16;
    const int aoff = half * 16;            // A: lanes<16 -> K {0..15}; lanes>=16 -> {16..31}
    const int bcol = c0 + l16;

    v8f c = {};
    #pragma unroll
    for (int kb = 0; kb < K; kb += 32) {
        const _Float16* xr = (kb < 128)
            ? X0h + (size_t)arow * FDIM + kb
            : X1h + (size_t)arow * FDIM + (kb - 128);
        v8h a0 = *(const v8h*)(xr + aoff);          // 16B contiguous
        v8h a1 = *(const v8h*)(xr + aoff + 8);      // 16B contiguous
        v16h a = __builtin_shufflevector(a0, a1,
                    0, 1, 2, 3, 4, 5, 6, 7, 8, 9, 10, 11, 12, 13, 14, 15);
        v16h b = *(const v16h*)(Wp + ((size_t)((kb >> 5) * 2 + half)) * 2048
                                   + (size_t)bcol * 16);   // 32B contiguous
        c = __builtin_amdgcn_wmma_f32_16x16x32_f16(false, a, false, b,
                                                   (short)0, c, false, false);
    }
    float bv = bias ? bias[bcol] : 0.0f;
    const int rbase = r0 + half * 8;       // C/D: VGPR v -> M = v (+8 for lanes>=16)
    #pragma unroll
    for (int v = 0; v < 8; ++v)
        Out[(size_t)(rbase + v) * FDIM + bcol] = c[v] + bv;
}

// ---- el[n,h] = <ft[n,h,:], al[h,:]> ; er likewise ----
__global__ void attn_dots(const float* __restrict__ ft,
                          const float* __restrict__ al,
                          const float* __restrict__ ar,
                          float* __restrict__ el, float* __restrict__ er)
{
    int t = blockIdx.x * blockDim.x + threadIdx.x;   // n*H + h
    if (t >= NNODES * HEADS) return;
    int h = t & 7, n = t >> 3;
    const float* fp  = ft + (size_t)n * FDIM + h * DHEAD;
    const float* alp = al + h * DHEAD;
    const float* arp = ar + h * DHEAD;
    float sl = 0.f, sr = 0.f;
    #pragma unroll
    for (int d = 0; d < DHEAD; ++d) { float f = fp[d]; sl += f * alp[d]; sr += f * arp[d]; }
    el[t] = sl; er[t] = sr;
}

__global__ void fill_u32(unsigned* __restrict__ p, unsigned v, int n) {
    int t = blockIdx.x * blockDim.x + threadIdx.x;
    if (t < n) p[t] = v;
}

__device__ __forceinline__ float edge_score(const float* el, const float* er,
                                            int s, int d, int h, float we) {
    float t = el[(size_t)s * HEADS + h] + er[(size_t)d * HEADS + h];
    t = t > 0.f ? t : 0.1f * t;     // leaky_relu(0.1)
    return we * t;
}

__global__ void edge_max(const int* __restrict__ src, const int* __restrict__ dst,
                         const float* __restrict__ w,
                         const float* __restrict__ el, const float* __restrict__ er,
                         unsigned* __restrict__ mo)
{
    int e = blockIdx.x * blockDim.x + threadIdx.x;
    if (e >= NEDGES) return;
    int s = src[e], d = dst[e];
    float we = w[e];
    unsigned* mp = mo + (size_t)d * HEADS;
    #pragma unroll
    for (int h = 0; h < HEADS; ++h)
        atomicMax(mp + h, ford(edge_score(el, er, s, d, h, we)));
}

__global__ void finalize_max(unsigned* __restrict__ mo, int n) {
    int t = blockIdx.x * blockDim.x + threadIdx.x;
    if (t >= n) return;
    unsigned u = mo[t];
    float m = (u == 0u) ? 0.f : funord(u);  // empty segment -> 0 (matches reference)
    ((float*)mo)[t] = m;
}

__global__ void edge_sum(const int* __restrict__ src, const int* __restrict__ dst,
                         const float* __restrict__ w,
                         const float* __restrict__ el, const float* __restrict__ er,
                         const float* __restrict__ m, float* __restrict__ ssum)
{
    int e = blockIdx.x * blockDim.x + threadIdx.x;
    if (e >= NEDGES) return;
    int s = src[e], d = dst[e];
    float we = w[e];
    #pragma unroll
    for (int h = 0; h < HEADS; ++h) {
        int di = d * HEADS + h;
        atomicAdd(ssum + di, __expf(edge_score(el, er, s, d, h, we) - m[di]));
    }
}

// one thread per (edge, head): 16-wide weighted scatter-add of ft[src] into acc[dst]
__global__ void edge_agg(const int* __restrict__ src, const int* __restrict__ dst,
                         const float* __restrict__ w,
                         const float* __restrict__ el, const float* __restrict__ er,
                         const float* __restrict__ m, const float* __restrict__ ssum,
                         const float* __restrict__ ft, float* __restrict__ acc)
{
    int t = blockIdx.x * blockDim.x + threadIdx.x;   // e*H + h
    if (t >= NEDGES * HEADS) return;
    int h = t & 7, e = t >> 3;
    int s = src[e], d = dst[e];
    int di = d * HEADS + h;
    float a = __expf(edge_score(el, er, s, d, h, w[e]) - m[di]) / ssum[di];
    const float4* fp = (const float4*)(ft + (size_t)s * FDIM + h * DHEAD);
    float* op = acc + (size_t)d * FDIM + h * DHEAD;
    #pragma unroll
    for (int q = 0; q < 4; ++q) {
        float4 f = fp[q];
        atomicAdd(op + q * 4 + 0, f.x * a);
        atomicAdd(op + q * 4 + 1, f.y * a);
        atomicAdd(op + q * 4 + 2, f.z * a);
        atomicAdd(op + q * 4 + 3, f.w * a);
    }
}

extern "C" void kernel_launch(void* const* d_in, const int* in_sizes, int n_in,
                              void* d_out, int out_size, void* d_ws, size_t ws_size,
                              hipStream_t stream) {
    const float* features = (const float*)d_in[0];
    const int*   src      = (const int*)  d_in[1];
    const int*   dst      = (const int*)  d_in[2];
    const float* w        = (const float*)d_in[3];
    const float* W1       = (const float*)d_in[4];
    const float* al1      = (const float*)d_in[5];
    const float* ar1      = (const float*)d_in[6];
    const float* W2       = (const float*)d_in[7];
    const float* al2      = (const float*)d_in[8];
    const float* ar2      = (const float*)d_in[9];
    const float* Wm       = (const float*)d_in[10];
    const float* bm       = (const float*)d_in[11];
    float* out = (float*)d_out;

    const size_t NF = (size_t)NNODES * FDIM;   // 12.8M
    const int    NH = NNODES * HEADS;          // 800k

    // workspace carve-up (~166 MB; everything hot is L2-resident)
    float*     ft   = (float*)d_ws;            // N*128 f32 : per-layer transformed feats
    float*     acc  = ft + NF;                 // N*128 f32 : aggregation accumulator
    float*     el   = acc + NF;                // N*H
    float*     er   = el + NH;                 // N*H
    unsigned*  mo   = (unsigned*)(er + NH);    // N*H (uint max -> float m in place)
    float*     mf   = (float*)mo;
    float*     ssum = (float*)(mo + NH);       // N*H (contiguous after mo)
    _Float16*  hA   = (_Float16*)(ssum + NH);  // N*128 f16 : features_h, later x2_h
    _Float16*  hB   = hA + NF;                 // N*128 f16 : x1_h
    _Float16*  Wp1  = hB + NF;                 // 128*128 packed f16
    _Float16*  Wp2  = Wp1 + (size_t)FDIM * FDIM;
    _Float16*  Wpm  = Wp2 + (size_t)FDIM * FDIM;  // 256*128 packed f16

    auto cdiv = [](int a, int b) { return (a + b - 1) / b; };

    // weight packing + feature conversion (tiny)
    pack_w<<<cdiv(FDIM * FDIM, 256), 256, 0, stream>>>(W1, Wp1, FDIM);
    pack_w<<<cdiv(FDIM * FDIM, 256), 256, 0, stream>>>(W2, Wp2, FDIM);
    pack_w<<<cdiv(2 * FDIM * FDIM, 256), 256, 0, stream>>>(Wm, Wpm, 2 * FDIM);
    cvt_f32_to_h<<<cdiv((int)NF, 256), 256, 0, stream>>>(features, hA, (int)NF);

    auto run_layer = [&](const _Float16* xh, const _Float16* Wp,
                         const float* al, const float* ar, _Float16* xout_h) {
        gemm_wmma_h<128><<<NNODES / 16, 256, 0, stream>>>(xh, xh, Wp, nullptr, ft);
        attn_dots<<<cdiv(NH, 256), 256, 0, stream>>>(ft, al, ar, el, er);
        fill_u32<<<cdiv(2 * NH, 256), 256, 0, stream>>>(mo, 0u, 2 * NH);   // mo + ssum
        fill_u32<<<cdiv((int)NF, 256), 256, 0, stream>>>((unsigned*)acc, 0u, (int)NF);
        edge_max<<<cdiv(NEDGES, 256), 256, 0, stream>>>(src, dst, w, el, er, mo);
        finalize_max<<<cdiv(NH, 256), 256, 0, stream>>>(mo, NH);
        edge_sum<<<cdiv(NEDGES, 256), 256, 0, stream>>>(src, dst, w, el, er, mf, ssum);
        edge_agg<<<cdiv(NEDGES * HEADS, 256), 256, 0, stream>>>(src, dst, w, el, er,
                                                                mf, ssum, ft, acc);
        relu_cvt<<<cdiv((int)NF, 256), 256, 0, stream>>>(acc, xout_h, (int)NF);
    };

    run_layer(hA, Wp1, al1, ar1, hB);   // layer 1: features_h -> x1_h (hB)
    run_layer(hB, Wp2, al2, ar2, hA);   // layer 2: x1_h -> x2_h (hA, reused)

    // out = [x1 | x2] @ Wm + bm   (K = 256)
    gemm_wmma_h<256><<<NNODES / 16, 256, 0, stream>>>(hB, hA, Wpm, bm, out);
}
